// MultiheadAttention_60816736911382
// MI455X (gfx1250) — matrix operand order repaired and checked by reading, not compiled, not run on hardware
//
#include <hip/hip_runtime.h>

typedef __bf16 bf16;
typedef __attribute__((ext_vector_type(16))) __bf16 v16bf;
typedef __attribute__((ext_vector_type(8)))  __bf16 v8bf;
typedef __attribute__((ext_vector_type(4)))  __bf16 v4bf;
typedef __attribute__((ext_vector_type(8)))  float  v8f;

#define BSZ 4
#define SEQ 2048
#define DIN 1024
#define EMB 1024
#define NH  16
#define HDIM 64
#define N3  3072   // 3*EMB

// ---------------- WMMA helpers (CDNA5 bf16, f32 accum) ----------------

__device__ __forceinline__ v8f wmma_bf16(v16bf a, v16bf b, v8f c) {
  return __builtin_amdgcn_wmma_f32_16x16x32_bf16(false, a, false, b,
                                                 (short)0, c, false, false);
}

// A-matrix 16x32 bf16 fragment: lanes 0-15 hold K {0..15},
// lanes 16-31 hold K {16..31}.  hi = lane>>4, p points at row (lane&15).
__device__ __forceinline__ v16bf load_a_frag(const bf16* p, int hi) {
  v8bf lo = *(const v8bf*)(p + hi * 16);
  v8bf hv = *(const v8bf*)(p + hi * 16 + 8);
  v16bf r;
#pragma unroll
  for (int i = 0; i < 8; ++i) { r[i] = lo[i]; r[i + 8] = hv[i]; }
  return r;
}

// B-matrix 32x16 fragment from a 32B-aligned source (global rows)
__device__ __forceinline__ v16bf load_b_frag(const bf16* p) {
  return *(const v16bf*)p;
}

// B-matrix fragment from 16B-aligned source (padded LDS rows): two b128 loads
__device__ __forceinline__ v16bf load_b_frag16(const bf16* p) {
  v8bf a = *(const v8bf*)p;
  v8bf b = *(const v8bf*)(p + 8);
  v16bf r;
#pragma unroll
  for (int i = 0; i < 8; ++i) { r[i] = a[i]; r[i + 8] = b[i]; }
  return r;
}

// ---------------- CDNA5 async global->LDS copy (ASYNCcnt path) ----------------
// GLOBAL_LOAD_ASYNC_TO_LDS_B128, GV mode: per-lane vdst = LDS byte address,
// per-lane vaddr = 64-bit global address.  Flat shared pointers carry the LDS
// offset in their low 32 bits (ISA 10.2: LDS aperture uses addr[31:0]).

__device__ __forceinline__ void async_copy_b128(const bf16* lds_dst, const bf16* gsrc) {
  unsigned int loff = (unsigned int)(unsigned long long)(uintptr_t)lds_dst;
  unsigned long long ga = (unsigned long long)(uintptr_t)gsrc;
  asm volatile("global_load_async_to_lds_b128 %0, %1, off"
               :: "v"(loff), "v"(ga) : "memory");
}

#define WAIT_ASYNCCNT(n) asm volatile("s_wait_asynccnt " #n ::: "memory")

// ---------------- fp32 -> bf16 conversion ----------------

__global__ void __launch_bounds__(256) cvt_bf16(const float* __restrict__ in,
                                                bf16* __restrict__ out, int n4) {
  int idx = blockIdx.x * 256 + threadIdx.x;
  if (idx < n4) {
    float4 v = ((const float4*)in)[idx];
    v4bf o;
    o[0] = (bf16)v.x; o[1] = (bf16)v.y; o[2] = (bf16)v.z; o[3] = (bf16)v.w;
    ((v4bf*)out)[idx] = o;
  }
}

// ---------------- GEMM1: qkv = X * Wqkv^T + b ----------------

__global__ void __launch_bounds__(256)
gemm_qkv(const bf16* __restrict__ Xb, const bf16* __restrict__ Wb,
         const float* __restrict__ bias,
         bf16* __restrict__ Qb, bf16* __restrict__ Kb, bf16* __restrict__ Vtb) {
  const int wave = blockIdx.x * 8 + (threadIdx.x >> 5);
  const int lane = threadIdx.x & 31;
  const int ln = lane & 15, hi = lane >> 4;
  const int tilesN = N3 / 64;                 // 48
  const int m0 = (wave / tilesN) * 16;
  const int c0 = (wave % tilesN) * 64;

  v8f acc[4];
#pragma unroll
  for (int t = 0; t < 4; ++t) {
    float bv = bias[c0 + t * 16 + ln];
#pragma unroll
    for (int j = 0; j < 8; ++j) acc[t][j] = bv;
  }

  const bf16* arow = Xb + (size_t)(m0 + ln) * DIN;
  for (int k0 = 0; k0 < DIN; k0 += 32) {
    __builtin_prefetch((const void*)(arow + k0 + 256), 0, 0);
    v16bf a = load_a_frag(arow + k0, hi);
    v16bf b[4];
#pragma unroll
    for (int t = 0; t < 4; ++t)
      b[t] = load_b_frag(Wb + (size_t)(c0 + t * 16 + ln) * DIN + k0 + hi * 16);
#pragma unroll
    for (int t = 0; t < 4; ++t)
      acc[t] = wmma_bf16(a, b[t], acc[t]);
  }

  // scatter: col -> (head, q/k/v, d); qkv reshaped [s, 16 heads, 192]
#pragma unroll
  for (int t = 0; t < 4; ++t) {
    int col  = c0 + t * 16 + ln;
    int head = col / 192;
    int r    = col % 192;
    int kind = r >> 6;
    int d    = r & 63;
#pragma unroll
    for (int j = 0; j < 8; ++j) {
      int row = m0 + hi * 8 + j;
      int b2 = row >> 11, s = row & 2047;
      bf16 v = (bf16)acc[t][j];
      size_t bh = (size_t)(b2 * NH + head);
      if (kind == 0)      Qb [(bh * SEQ + s) * HDIM + d] = v;
      else if (kind == 1) Kb [(bh * SEQ + s) * HDIM + d] = v;
      else                Vtb[(bh * HDIM + d) * SEQ + s] = v;
    }
  }
}

// ---------------- Flash attention ----------------
// 4 waves/block share one (b,h).  K/V chunks are staged into LDS with
// global_load_async_to_lds_b128, double-buffered so chunk i+1's DMA overlaps
// chunk i's WMMA compute; in-order async completion lets s_wait_asynccnt 4
// retire the current chunk while the next stays in flight.

#define KPAD 72   // 64 + 8 bf16 row pad for K chunk
#define VPAD 40   // 32 + 8 bf16 row pad for V chunk

__device__ __forceinline__ void stage_chunk(bf16* KcB, bf16* VcB,
                                            const bf16* Kbase, const bf16* Vbase,
                                            int kk, int tid) {
  // K chunk [32 x 64]: 256 x 16B, 2 per thread
  int r0 = tid >> 3,         col0 = (tid & 7) * 8;
  int r1 = (tid + 128) >> 3, col1 = ((tid + 128) & 7) * 8;
  async_copy_b128(&KcB[r0 * KPAD + col0], Kbase + (size_t)(kk + r0) * HDIM + col0);
  async_copy_b128(&KcB[r1 * KPAD + col1], Kbase + (size_t)(kk + r1) * HDIM + col1);
  // V chunk [64 x 32] from Vt rows: 256 x 16B, 2 per thread
  int d0 = tid >> 2,         sc0 = (tid & 3) * 8;
  int d1 = (tid + 128) >> 2, sc1 = ((tid + 128) & 3) * 8;
  async_copy_b128(&VcB[d0 * VPAD + sc0], Vbase + (size_t)d0 * SEQ + kk + sc0);
  async_copy_b128(&VcB[d1 * VPAD + sc1], Vbase + (size_t)d1 * SEQ + kk + sc1);
}

__global__ void __launch_bounds__(128)
attn_fwd(const bf16* __restrict__ Qb, const bf16* __restrict__ Kb,
         const bf16* __restrict__ Vtb, bf16* __restrict__ Vals) {
  __shared__ __align__(16) bf16 KcS[2][32 * KPAD];
  __shared__ __align__(16) bf16 VcS[2][64 * VPAD];
  __shared__ __align__(32) bf16 ldsP[4][16 * 32];

  const int tid  = threadIdx.x;
  const int widx = tid >> 5;
  const int lane = tid & 31;
  const int ln = lane & 15, hi = lane >> 4;
  const int bh = blockIdx.x >> 5;                     // 32 blocks per (b,h)
  const int s0 = (((blockIdx.x & 31) << 2) + widx) * 16;

  const bf16* qrow = Qb + ((size_t)bh * SEQ + s0 + ln) * HDIM;
  v16bf aq0 = load_a_frag(qrow + 0, hi);
  v16bf aq1 = load_a_frag(qrow + 32, hi);

  float mrow[8], lrow[8];
#pragma unroll
  for (int j = 0; j < 8; ++j) { mrow[j] = -1e30f; lrow[j] = 0.0f; }
  v8f o[4];
#pragma unroll
  for (int t = 0; t < 4; ++t)
#pragma unroll
    for (int j = 0; j < 8; ++j) o[t][j] = 0.0f;

  bf16* myP = ldsP[widx];
  const bf16* Kbase = Kb  + (size_t)bh * SEQ * HDIM;
  const bf16* Vbase = Vtb + (size_t)bh * HDIM * SEQ;
  const float scale = 0.125f;                         // 1/sqrt(64)

  // prologue: stage chunk 0 into buffer 0
  stage_chunk(KcS[0], VcS[0], Kbase, Vbase, 0, tid);
  int buf = 0;

  for (int kk = 0; kk < SEQ; kk += 32) {
    if (kk + 32 < SEQ) {
      // overlap: issue next chunk's DMA, then retire only the current chunk
      stage_chunk(KcS[buf ^ 1], VcS[buf ^ 1], Kbase, Vbase, kk + 32, tid);
      WAIT_ASYNCCNT(4);
    } else {
      WAIT_ASYNCCNT(0);
    }
    __syncthreads();

    const bf16* Kc = KcS[buf];
    const bf16* Vc = VcS[buf];

    // ---- logits tile 16x32 = two 16x16 C tiles, K frags from LDS ----
    v8f st[2];
#pragma unroll
    for (int t = 0; t < 2; ++t) {
#pragma unroll
      for (int j = 0; j < 8; ++j) st[t][j] = 0.0f;
      const bf16* kr = &Kc[(t * 16 + ln) * KPAD];
      st[t] = wmma_bf16(aq0, load_b_frag16(kr + 0  + hi * 16), st[t]);
      st[t] = wmma_bf16(aq1, load_b_frag16(kr + 32 + hi * 16), st[t]);
    }

    // ---- online softmax (rows in VGPR index, cols across 16 lanes) ----
    float alpha[8];
#pragma unroll
    for (int j = 0; j < 8; ++j) {
      float v0 = st[0][j] * scale, v1 = st[1][j] * scale;
      float mx = fmaxf(v0, v1);
#pragma unroll
      for (int off = 1; off < 16; off <<= 1)
        mx = fmaxf(mx, __shfl_xor(mx, off, 32));
      float mnew = fmaxf(mrow[j], mx);
      alpha[j] = __expf(mrow[j] - mnew);
      float p0 = __expf(v0 - mnew), p1 = __expf(v1 - mnew);
      st[0][j] = p0; st[1][j] = p1;
      float rs = p0 + p1;
#pragma unroll
      for (int off = 1; off < 16; off <<= 1)
        rs += __shfl_xor(rs, off, 32);
      lrow[j] = lrow[j] * alpha[j] + rs;
      mrow[j] = mnew;
    }

    // ---- transpose P (C layout -> A layout) through per-wave LDS tile ----
#pragma unroll
    for (int t = 0; t < 2; ++t)
#pragma unroll
      for (int j = 0; j < 8; ++j)
        myP[(hi * 8 + j) * 32 + t * 16 + ln] = (bf16)st[t][j];
    v16bf pa = load_a_frag(myP + ln * 32, hi);

    // ---- O = O*alpha + P(16x32) * V(32x64), V frags from LDS ----
#pragma unroll
    for (int t = 0; t < 4; ++t) {
      const bf16* vr = &Vc[(t * 16 + ln) * VPAD + hi * 16];
      v8f c = o[t];
#pragma unroll
      for (int j = 0; j < 8; ++j) c[j] *= alpha[j];
      o[t] = wmma_bf16(pa, load_b_frag16(vr), c);
    }
    __syncthreads();   // all waves done reading before buf^1 is re-staged
    buf ^= 1;
  }

  const int b = bh / NH, h = bh % NH;
#pragma unroll
  for (int t = 0; t < 4; ++t) {
    int d = t * 16 + ln;
#pragma unroll
    for (int j = 0; j < 8; ++j) {
      int s = s0 + hi * 8 + j;
      float v = o[t][j] / lrow[j];
      Vals[((size_t)b * SEQ + s) * EMB + h * HDIM + d] = (bf16)v;
    }
  }
}

// ---------------- GEMM2: out = vals * Wo^T + b_o (fp32 out) ----------------

__global__ void __launch_bounds__(256)
gemm_out(const bf16* __restrict__ Vb, const bf16* __restrict__ Wb,
         const float* __restrict__ bias, float* __restrict__ Out) {
  const int wave = blockIdx.x * 8 + (threadIdx.x >> 5);
  const int lane = threadIdx.x & 31;
  const int ln = lane & 15, hi = lane >> 4;
  const int tilesN = EMB / 64;
  const int m0 = (wave / tilesN) * 16;
  const int c0 = (wave % tilesN) * 64;

  v8f acc[4];
#pragma unroll
  for (int t = 0; t < 4; ++t) {
    float bv = bias[c0 + t * 16 + ln];
#pragma unroll
    for (int j = 0; j < 8; ++j) acc[t][j] = bv;
  }

  const bf16* arow = Vb + (size_t)(m0 + ln) * EMB;
  for (int k0 = 0; k0 < EMB; k0 += 32) {
    __builtin_prefetch((const void*)(arow + k0 + 256), 0, 0);
    v16bf a = load_a_frag(arow + k0, hi);
    v16bf b[4];
#pragma unroll
    for (int t = 0; t < 4; ++t)
      b[t] = load_b_frag(Wb + (size_t)(c0 + t * 16 + ln) * EMB + k0 + hi * 16);
#pragma unroll
    for (int t = 0; t < 4; ++t)
      acc[t] = wmma_bf16(a, b[t], acc[t]);
  }

#pragma unroll
  for (int t = 0; t < 4; ++t) {
    int col = c0 + t * 16 + ln;
#pragma unroll
    for (int j = 0; j < 8; ++j) {
      int row = m0 + hi * 8 + j;
      Out[(size_t)row * EMB + col] = acc[t][j];
    }
  }
}

// ---------------- launcher ----------------

extern "C" void kernel_launch(void* const* d_in, const int* in_sizes, int n_in,
                              void* d_out, int out_size, void* d_ws, size_t ws_size,
                              hipStream_t stream) {
  (void)in_sizes; (void)n_in; (void)out_size; (void)ws_size;
  const float* X    = (const float*)d_in[0];
  const float* Wqkv = (const float*)d_in[1];
  const float* bqkv = (const float*)d_in[2];
  const float* Wo   = (const float*)d_in[3];
  const float* bo   = (const float*)d_in[4];
  float* Out = (float*)d_out;

  char* ws = (char*)d_ws;
  size_t off = 0;
  bf16* Xb    = (bf16*)(ws + off); off += (size_t)BSZ * SEQ * DIN * 2;
  bf16* Wqkvb = (bf16*)(ws + off); off += (size_t)N3 * DIN * 2;
  bf16* Wob   = (bf16*)(ws + off); off += (size_t)EMB * EMB * 2;
  bf16* Qb    = (bf16*)(ws + off); off += (size_t)BSZ * NH * SEQ * HDIM * 2;
  bf16* Kb    = (bf16*)(ws + off); off += (size_t)BSZ * NH * SEQ * HDIM * 2;
  bf16* Vtb   = (bf16*)(ws + off); off += (size_t)BSZ * NH * SEQ * HDIM * 2;
  bf16* Vals  = (bf16*)(ws + off); off += (size_t)BSZ * SEQ * EMB * 2;

  cvt_bf16<<<(BSZ * SEQ * DIN) / 4 / 256, 256, 0, stream>>>(X, Xb, (BSZ * SEQ * DIN) / 4);
  cvt_bf16<<<(N3 * DIN) / 4 / 256, 256, 0, stream>>>(Wqkv, Wqkvb, (N3 * DIN) / 4);
  cvt_bf16<<<(EMB * EMB) / 4 / 256, 256, 0, stream>>>(Wo, Wob, (EMB * EMB) / 4);

  gemm_qkv<<<3072, 256, 0, stream>>>(Xb, Wqkvb, bqkv, Qb, Kb, Vtb);
  attn_fwd<<<2048, 128, 0, stream>>>(Qb, Kb, Vtb, Vals);
  gemm_out<<<1024, 256, 0, stream>>>(Vals, Wob, bo, Out);
}
